// DNM_25383256719840
// MI455X (gfx1250) — matrix operand order repaired and checked by reading, not compiled, not run on hardware
//
#include <hip/hip_runtime.h>
#include <hip/hip_bf16.h>
#include <math.h>

#define B_ 64
#define S_ 512
#define H_ 256

typedef __attribute__((ext_vector_type(16))) __bf16 v16bf;
typedef __attribute__((ext_vector_type(8)))  float  v8f;

union Frag {
    v16bf v;
    unsigned int u[8];
    unsigned short h[16];
};

__device__ __forceinline__ unsigned short f2bf(float x) {
    union { float f; unsigned u; } a; a.f = x;
    unsigned r = a.u + 0x7FFFu + ((a.u >> 16) & 1u);
    return (unsigned short)(r >> 16);
}
__device__ __forceinline__ float bf2f(unsigned short h) {
    union { unsigned u; float f; } a; a.u = ((unsigned)h) << 16;
    return a.f;
}
__device__ __forceinline__ float sigmoidf_(float x) { return 1.0f / (1.0f + __expf(-x)); }

__device__ __forceinline__ v8f wmma_bf16(const Frag& a, const Frag& b, v8f c) {
    return __builtin_amdgcn_wmma_f32_16x16x32_bf16(
        false, a.v, false, b.v, (short)0, c, false, false);
}

// A fragment (16x32 bf16): lanes 0-15 row M=lane K=kbase..+7 & kbase+16..+23 (kbase=kt*32),
// lanes 16-31 same rows, kbase+8. Source row is contiguous bf16 -> two b128 loads.
__device__ __forceinline__ void load_afrag_g(Frag& f, const unsigned short* row, int kbase) {
    uint4 a = *(const uint4*)(row + kbase);
    uint4 b = *(const uint4*)(row + kbase + 16);
    f.u[0]=a.x; f.u[1]=a.y; f.u[2]=a.z; f.u[3]=a.w;
    f.u[4]=b.x; f.u[5]=b.y; f.u[6]=b.z; f.u[7]=b.w;
}

// B fragments pre-swizzled: packed[((kt*NT+nt)*32+lane)*16 + e]
__device__ __forceinline__ void load_bfrag(Frag& f, const unsigned short* packed,
                                           int NT, int kt, int nt, int lane) {
    const uint4* p = (const uint4*)(packed + (((size_t)(kt * NT + nt)) * 32 + lane) * 16);
    uint4 a = p[0], b = p[1];
    f.u[0]=a.x; f.u[1]=a.y; f.u[2]=a.z; f.u[3]=a.w;
    f.u[4]=b.x; f.u[5]=b.y; f.u[6]=b.z; f.u[7]=b.w;
}

__device__ __forceinline__ void load_afrag_lds(Frag& f, const unsigned short* row, int kbase) {
    uint4 x = *(const uint4*)(row + kbase);
    uint4 y = *(const uint4*)(row + kbase + 16);
    f.u[0]=x.x; f.u[1]=x.y; f.u[2]=x.z; f.u[3]=x.w;
    f.u[4]=y.x; f.u[5]=y.y; f.u[6]=y.z; f.u[7]=y.w;
}

// 48KB slab copy, global -> LDS, async (ASYNCcnt); 12 b128 copies per wave,
// 256 threads cover 16 rows x 768 f32.
__device__ __forceinline__ void async_slab_load(const float* gsrc, unsigned ldsbase, int tid) {
#pragma unroll
    for (int it = 0; it < 12; ++it) {
        int i = it * 256 + tid;   // uint4 index
        const void* ga = (const void*)((const char*)gsrc + (size_t)i * 16);
        unsigned la = ldsbase + (unsigned)i * 16;
        asm volatile("global_load_async_to_lds_b128 %0, %1, off"
                     :: "v"(la), "v"(ga)
                     : "memory");
    }
}

// ---------- staging kernels ----------

__global__ void cvt_bf16_kernel(const float* __restrict__ src, unsigned short* __restrict__ dst, int n) {
    int i = blockIdx.x * blockDim.x + threadIdx.x;
    if (i < n) dst[i] = f2bf(src[i]);
}

// Pack W [Ntot, Ktot] (row-major) into WMMA B-operand fragments (B[k][n] = W[n][k]).
// B 32x16 bf16 layout: lane n (0-15) = col n, elems 0..7 = K 0..7, 8..15 = K 16..23; lane n+16 = col n, K 8..15 & 24..31.
__global__ void pack_b_kernel(const float* __restrict__ W, unsigned short* __restrict__ out,
                              int Ktot, int Ntot) {
    int idx = blockIdx.x * blockDim.x + threadIdx.x;
    int KT = Ktot >> 5, NT = Ntot >> 4;
    int total = KT * NT * 512;
    if (idx >= total) return;
    int e    = idx & 15;
    int lane = (idx >> 4) & 31;
    int tile = idx >> 9;
    int nt = tile % NT, kt = tile / NT;
    int k = kt * 32 + (e >> 3) * 16 + ((lane >> 4) & 1) * 8 + (e & 7);
    int n = nt * 16 + (lane & 15);
    out[idx] = f2bf(W[(size_t)n * Ktot + k]);
}

// ---------- one-time: gi[s][b][0:768] = f[b,s] @ gate_Wih.T + bih ----------

__global__ void __launch_bounds__(256)
gi_gemm_kernel(const unsigned short* __restrict__ fB, const unsigned short* __restrict__ Wp,
               const float* __restrict__ bih, float* __restrict__ gi) {
    int blk = blockIdx.x;             // 64 * 32
    int b  = blk >> 5;
    int s0 = (blk & 31) << 4;
    int tid = threadIdx.x, lane = tid & 31, wv = tid >> 5;
    int ln15 = lane & 15, hi = (lane >> 4) & 1;
    const unsigned short* arow = fB + ((size_t)(b * S_ + s0 + ln15)) * H_;
    v8f acc[6] = {};
    for (int kt = 0; kt < 8; ++kt) {
        Frag a, bf[6];
        load_afrag_g(a, arow, kt * 32 + hi * 8);
#pragma unroll
        for (int t = 0; t < 6; ++t) load_bfrag(bf[t], Wp, 48, kt, wv * 6 + t, lane);
#pragma unroll
        for (int t = 0; t < 6; ++t) acc[t] = wmma_bf16(a, bf[t], acc[t]);
    }
#pragma unroll
    for (int t = 0; t < 6; ++t) {
        int n = (wv * 6 + t) * 16 + ln15;
        float bias = bih[n];
#pragma unroll
        for (int v = 0; v < 8; ++v) {
            int srow = s0 + v + 8 * hi;
            gi[((size_t)srow * B_ + b) * 768 + n] = acc[t][v] + bias;
        }
    }
}

// ---------- per pass: attention logits via on-the-fly z features ----------

__global__ void __launch_bounds__(256)
attn_kernel(const unsigned short* __restrict__ fB, const unsigned short* __restrict__ qB,
            const unsigned short* __restrict__ mB, const unsigned short* __restrict__ g1p,
            const float* __restrict__ g1b, const float* __restrict__ g2W,
            const float* __restrict__ g2b, float* __restrict__ logits) {
    __shared__ float red[16];
    int blk = blockIdx.x;
    int b  = blk >> 5;
    int s0 = (blk & 31) << 4;
    int tid = threadIdx.x, lane = tid & 31, wv = tid >> 5;
    int ln15 = lane & 15, hi = (lane >> 4) & 1;
    if (tid < 16) red[tid] = 0.0f;
    __syncthreads();
    const unsigned short* frow = fB + ((size_t)(b * S_ + s0 + ln15)) * H_;
    const unsigned short* qrow = qB + (size_t)b * H_;
    const unsigned short* mrow = mB + (size_t)b * H_;
    v8f acc[2] = {};
    for (int kt = 0; kt < 56; ++kt) {
        // pull next k-tile's B fragments toward the WGP while we compute
        if (kt + 1 < 56) {
            __builtin_prefetch(g1p + (((size_t)((kt + 1) * 16 + wv * 2)) * 32 + lane) * 16, 0, 3);
        }
        int c = kt >> 3;                        // which of 7 z-components
        int kbase = (kt & 7) * 32 + hi * 8;
        Frag a;
        if (c == 1) {
            load_afrag_g(a, mrow, kbase);
        } else if (c == 2) {
            load_afrag_g(a, qrow, kbase);
        } else if (c == 0) {
            load_afrag_g(a, frow, kbase);
        } else {
            Frag ff, of;
            load_afrag_g(ff, frow, kbase);
            load_afrag_g(of, (c == 3 || c == 5) ? qrow : mrow, kbase);
            bool isMul = (c < 5);
#pragma unroll
            for (int i = 0; i < 16; ++i) {
                float x = bf2f(ff.h[i]), y = bf2f(of.h[i]);
                a.h[i] = f2bf(isMul ? x * y : fabsf(x - y));
            }
        }
        Frag bf[2];
#pragma unroll
        for (int t = 0; t < 2; ++t) load_bfrag(bf[t], g1p, 16, kt, wv * 2 + t, lane);
#pragma unroll
        for (int t = 0; t < 2; ++t) acc[t] = wmma_bf16(a, bf[t], acc[t]);
    }
#pragma unroll
    for (int t = 0; t < 2; ++t) {
        int n = (wv * 2 + t) * 16 + ln15;
        float bias = g1b[n], w2 = g2W[n];
#pragma unroll
        for (int v = 0; v < 8; ++v) {
            atomicAdd(&red[v + 8 * hi], tanhf(acc[t][v] + bias) * w2);
        }
    }
    __syncthreads();
    if (tid < 16) logits[(size_t)b * S_ + s0 + tid] = red[tid] + g2b[0];
}

__global__ void __launch_bounds__(256)
softmax_kernel(const float* __restrict__ logits, float* __restrict__ g) {
    __shared__ float sh[256];
    int b = blockIdx.x, tid = threadIdx.x;
    const float* row = logits + (size_t)b * S_;
    float a0 = row[tid], a1 = row[tid + 256];
    sh[tid] = fmaxf(a0, a1);
    __syncthreads();
    for (int st = 128; st > 0; st >>= 1) {
        if (tid < st) sh[tid] = fmaxf(sh[tid], sh[tid + st]);
        __syncthreads();
    }
    float mx = sh[0];
    __syncthreads();
    float e0 = __expf(a0 - mx), e1 = __expf(a1 - mx);
    sh[tid] = e0 + e1;
    __syncthreads();
    for (int st = 128; st > 0; st >>= 1) {
        if (tid < st) sh[tid] += sh[tid + st];
        __syncthreads();
    }
    float inv = 1.0f / sh[0];
    g[(size_t)b * S_ + tid]       = e0 * inv;
    g[(size_t)b * S_ + tid + 256] = e1 * inv;
}

// ---------- per pass: 512-step gated GRU scan; 16 batch rows per WG, h in LDS ----------
// wave w owns N-tiles {2w,2w+1, 16+2w,17+2w, 32+2w,33+2w} so its (r,z,n) gate columns
// for h-cols [32w,32w+32) live entirely in its own accumulators.
// All 48 B fragments (step-invariant) are pinned in VGPRs for the whole 512-step
// recurrence; the per-step gi gate slab (48KB) is double-buffered into LDS with
// global_load_async_to_lds_b128 under the WMMAs (ASYNCcnt pipelining).
// Dynamic LDS: hF 16KB | hB 8KB | gi slabs 2x48KB = 120KB total.

__global__ void __launch_bounds__(256, 1)
scan_kernel(const unsigned short* __restrict__ Whhp, const float* __restrict__ bhh,
            const float* __restrict__ gi, const float* __restrict__ g,
            const int* __restrict__ lengths, float* __restrict__ e_out) {
    extern __shared__ __align__(16) char smem[];
    float*          hF   = (float*)smem;                          // [16][256]
    unsigned short* hB   = (unsigned short*)(smem + 16384);       // [16][256]
    float*          gbuf = (float*)(smem + 16384 + 8192);         // [2][16*768]

    int b0 = blockIdx.x * 16;
    int tid = threadIdx.x, lane = tid & 31, wv = tid >> 5;
    int ln15 = lane & 15, hi = (lane >> 4) & 1;

    for (int i = tid; i < 16 * 256; i += 256) {
        hF[i] = 0.0f;
        hB[i] = 0;
    }
    int ntl[6];
#pragma unroll
    for (int t = 0; t < 6; ++t) ntl[t] = (t >> 1) * 16 + 2 * wv + (t & 1);
    int cb0 = 32 * wv + ln15, cb1 = 32 * wv + 16 + ln15;
    float bhhv[6];
    bhhv[0] = bhh[cb0];       bhhv[1] = bhh[cb1];
    bhhv[2] = bhh[256 + cb0]; bhhv[3] = bhh[256 + cb1];
    bhhv[4] = bhh[512 + cb0]; bhhv[5] = bhh[512 + cb1];
    int lenv[8];
#pragma unroll
    for (int v = 0; v < 8; ++v) lenv[v] = lengths[b0 + v + 8 * hi];

    // pin all step-invariant B fragments in registers (48 frags = 384 VGPRs)
    Frag bw[48];
#pragma unroll
    for (int kt = 0; kt < 8; ++kt) {
#pragma unroll
        for (int t = 0; t < 6; ++t) {
            load_bfrag(bw[kt * 6 + t], Whhp, 48, kt, ntl[t], lane);
        }
    }

    // kick off async staging of the s=0 gi slab into buffer 0
    async_slab_load(gi + (size_t)b0 * 768, (unsigned)(size_t)(void*)gbuf, tid);
    __syncthreads();

    for (int s = 0; s < S_; ++s) {
        int cur = s & 1;
        // stage next step's slab under this step's WMMAs
        if (s + 1 < S_) {
            async_slab_load(gi + ((size_t)(s + 1) * B_ + b0) * 768,
                            (unsigned)(size_t)(void*)(gbuf + (cur ^ 1) * 12288), tid);
        }
        v8f acc[6] = {};
#pragma unroll
        for (int kt = 0; kt < 8; ++kt) {
            Frag a;
            load_afrag_lds(a, &hB[ln15 * 256], kt * 32 + hi * 8);
#pragma unroll
            for (int t = 0; t < 6; ++t) acc[t] = wmma_bf16(a, bw[kt * 6 + t], acc[t]);
        }
        // slab for step s complete: <=12 newest (step s+1) may remain outstanding
        if (s + 1 < S_) asm volatile("s_wait_asynccnt 12" ::: "memory");
        else            asm volatile("s_wait_asynccnt 0"  ::: "memory");
        __syncthreads();   // hB reads done + async LDS writes visible to all waves
        float gjv[8];
#pragma unroll
        for (int v = 0; v < 8; ++v) gjv[v] = g[(size_t)(b0 + v + 8 * hi) * S_ + s];
        const float* slab = gbuf + cur * 12288;
#pragma unroll
        for (int t2 = 0; t2 < 2; ++t2) {
            int c = (t2 == 0) ? cb0 : cb1;
#pragma unroll
            for (int v = 0; v < 8; ++v) {
                int M = v + 8 * hi;
                const float* gp = slab + M * 768;
                float r  = sigmoidf_(gp[c]       + acc[t2][v]     + bhhv[t2]);
                float z  = sigmoidf_(gp[256 + c] + acc[2 + t2][v] + bhhv[2 + t2]);
                float nn = tanhf    (gp[512 + c] + r * (acc[4 + t2][v] + bhhv[4 + t2]));
                float h  = hF[M * 256 + c];
                float hn2   = (1.0f - z) * nn + z * h;
                float gj    = gjv[v];
                float gated = gj * hn2 + (1.0f - gj) * h;
                float out   = (s < lenv[v]) ? gated : h;
                hF[M * 256 + c] = out;
                hB[M * 256 + c] = f2bf(out);
            }
        }
        __syncthreads();   // h updates visible before next step's A-frag reads
    }
#pragma unroll
    for (int t2 = 0; t2 < 2; ++t2) {
        int c = (t2 == 0) ? cb0 : cb1;
#pragma unroll
        for (int v = 0; v < 8; ++v) {
            int M = v + 8 * hi;
            e_out[(size_t)(b0 + M) * H_ + c] = hF[M * 256 + c];
        }
    }
}

// ---------- per pass: m = GRUcell(e, m, mem_*) ----------

__global__ void __launch_bounds__(256)
memgru_kernel(const float* __restrict__ eF, const float* __restrict__ mF,
              const unsigned short* __restrict__ Wip, const unsigned short* __restrict__ Whp,
              const float* __restrict__ bih, const float* __restrict__ bhh,
              float* __restrict__ m_next) {
    __shared__ __align__(16) unsigned short eB[16][256];
    __shared__ __align__(16) unsigned short mB[16][256];
    int b0 = blockIdx.x * 16;
    int tid = threadIdx.x, lane = tid & 31, wv = tid >> 5;
    int ln15 = lane & 15, hi = (lane >> 4) & 1;
    for (int i = tid; i < 4096; i += 256) {
        int r = i >> 8, c = i & 255;
        eB[r][c] = f2bf(eF[(size_t)(b0 + r) * H_ + c]);
        mB[r][c] = f2bf(mF[(size_t)(b0 + r) * H_ + c]);
    }
    __syncthreads();
    int ntl[6];
#pragma unroll
    for (int t = 0; t < 6; ++t) ntl[t] = (t >> 1) * 16 + 2 * wv + (t & 1);
    v8f aci[6] = {}, ach[6] = {};
#pragma unroll
    for (int kt = 0; kt < 8; ++kt) {
        int kbase = kt * 32 + hi * 8;
        Frag ae, am;
        load_afrag_lds(ae, &eB[ln15][0], kbase);
        load_afrag_lds(am, &mB[ln15][0], kbase);
        Frag bi[6];
#pragma unroll
        for (int t = 0; t < 6; ++t) load_bfrag(bi[t], Wip, 48, kt, ntl[t], lane);
#pragma unroll
        for (int t = 0; t < 6; ++t) aci[t] = wmma_bf16(ae, bi[t], aci[t]);
        Frag bh[6];
#pragma unroll
        for (int t = 0; t < 6; ++t) load_bfrag(bh[t], Whp, 48, kt, ntl[t], lane);
#pragma unroll
        for (int t = 0; t < 6; ++t) ach[t] = wmma_bf16(am, bh[t], ach[t]);
    }
#pragma unroll
    for (int t2 = 0; t2 < 2; ++t2) {
        int c = 32 * wv + 16 * t2 + ln15;
#pragma unroll
        for (int v = 0; v < 8; ++v) {
            int brow = b0 + v + 8 * hi;
            float r  = sigmoidf_((aci[t2][v]     + bih[c])       + (ach[t2][v]     + bhh[c]));
            float z  = sigmoidf_((aci[2 + t2][v] + bih[256 + c]) + (ach[2 + t2][v] + bhh[256 + c]));
            float nn = tanhf    ((aci[4 + t2][v] + bih[512 + c]) + r * (ach[4 + t2][v] + bhh[512 + c]));
            float mo = mF[(size_t)brow * H_ + c];
            m_next[(size_t)brow * H_ + c] = (1.0f - z) * nn + z * mo;
        }
    }
}

// ---------- host ----------

extern "C" void kernel_launch(void* const* d_in, const int* in_sizes, int n_in,
                              void* d_out, int out_size, void* d_ws, size_t ws_size,
                              hipStream_t stream) {
    (void)in_sizes; (void)n_in; (void)out_size; (void)ws_size;
    const float* f    = (const float*)d_in[0];
    const float* q    = (const float*)d_in[1];
    const int*   len  = (const int*)  d_in[2];
    const float* gWih = (const float*)d_in[3];
    const float* gWhh = (const float*)d_in[4];
    const float* gbih = (const float*)d_in[5];
    const float* gbhh = (const float*)d_in[6];
    const float* mWih = (const float*)d_in[7];
    const float* mWhh = (const float*)d_in[8];
    const float* mbih = (const float*)d_in[9];
    const float* mbhh = (const float*)d_in[10];
    const float* g1W  = (const float*)d_in[11];
    const float* g1b  = (const float*)d_in[12];
    const float* g2W  = (const float*)d_in[13];
    const float* g2b  = (const float*)d_in[14];

    char* ws = (char*)d_ws;
    size_t off = 0;
    auto alloc = [&](size_t bytes) -> void* {
        void* p = ws + off;
        off = (off + bytes + 255) & ~((size_t)255);
        return p;
    };
    unsigned short* fB    = (unsigned short*)alloc((size_t)B_ * S_ * H_ * 2);
    unsigned short* qB    = (unsigned short*)alloc((size_t)B_ * H_ * 2);
    unsigned short* mBh   = (unsigned short*)alloc((size_t)B_ * H_ * 2);
    unsigned short* gWihP = (unsigned short*)alloc((size_t)8 * 48 * 512 * 2);
    unsigned short* gWhhP = (unsigned short*)alloc((size_t)8 * 48 * 512 * 2);
    unsigned short* mWihP = (unsigned short*)alloc((size_t)8 * 48 * 512 * 2);
    unsigned short* mWhhP = (unsigned short*)alloc((size_t)8 * 48 * 512 * 2);
    unsigned short* g1P   = (unsigned short*)alloc((size_t)56 * 16 * 512 * 2);
    float* gi     = (float*)alloc((size_t)S_ * B_ * 768 * 4);
    float* logits = (float*)alloc((size_t)B_ * S_ * 4);
    float* gsm    = (float*)alloc((size_t)B_ * S_ * 4);
    float* eF     = (float*)alloc((size_t)B_ * H_ * 4);
    float* m0     = (float*)alloc((size_t)B_ * H_ * 4);
    float* m1     = (float*)alloc((size_t)B_ * H_ * 4);

    // one-time staging (pass-invariant)
    int nf = B_ * S_ * H_;
    cvt_bf16_kernel<<<(nf + 255) / 256, 256, 0, stream>>>(f, fB, nf);
    cvt_bf16_kernel<<<64, 256, 0, stream>>>(q, qB, B_ * H_);
    pack_b_kernel<<<(8 * 48 * 512 + 255) / 256, 256, 0, stream>>>(gWih, gWihP, 256, 768);
    pack_b_kernel<<<(8 * 48 * 512 + 255) / 256, 256, 0, stream>>>(gWhh, gWhhP, 256, 768);
    pack_b_kernel<<<(8 * 48 * 512 + 255) / 256, 256, 0, stream>>>(mWih, mWihP, 256, 768);
    pack_b_kernel<<<(8 * 48 * 512 + 255) / 256, 256, 0, stream>>>(mWhh, mWhhP, 256, 768);
    pack_b_kernel<<<(56 * 16 * 512 + 255) / 256, 256, 0, stream>>>(g1W, g1P, 1792, 256);
    gi_gemm_kernel<<<B_ * (S_ / 16), 256, 0, stream>>>(fB, gWihP, gbih, gi);
    hipMemcpyAsync(m0, q, (size_t)B_ * H_ * 4, hipMemcpyDeviceToDevice, stream);

    float* mcur = m0;
    float* mnext = m1;
    for (int p = 0; p < 5; ++p) {
        cvt_bf16_kernel<<<64, 256, 0, stream>>>(mcur, mBh, B_ * H_);
        attn_kernel<<<B_ * (S_ / 16), 256, 0, stream>>>(fB, qB, mBh, g1P, g1b, g2W, g2b, logits);
        softmax_kernel<<<B_, 256, 0, stream>>>(logits, gsm);
        scan_kernel<<<B_ / 16, 256, 122880, stream>>>(gWhhP, gbhh, gi, gsm, len, eF);
        memgru_kernel<<<B_ / 16, 256, 0, stream>>>(eF, mcur, mWihP, mWhhP, mbih, mbhh, mnext);
        float* t = mcur; mcur = mnext; mnext = t;
    }
    hipMemcpyAsync(d_out, mcur, (size_t)B_ * H_ * 4, hipMemcpyDeviceToDevice, stream);
}
